// NeighborhoodAttention2D_85117661872430
// MI455X (gfx1250) — matrix operand order repaired and checked by reading, not compiled, not run on hardware
//
#include <hip/hip_runtime.h>
#include <hip/hip_bf16.h>

#define BB      8
#define CC      256
#define HH      96
#define WWD     96
#define NHEADS  8
#define HD      32
#define KWIN    7
#define HWSZ    (HH*WWD)          // 9216
#define NTOK    (BB*HWSZ)         // 73728
#define SCLIP   (HH-KWIN)         // 89

typedef __attribute__((ext_vector_type(16))) __bf16 v16bf;
typedef __attribute__((ext_vector_type(8)))  __bf16 v8bf;
typedef v8bf __attribute__((aligned(2)))            v8bf_u;   // reduced-alignment loads
typedef __attribute__((ext_vector_type(8)))  float  v8f;

__device__ __forceinline__ v16bf cat16(v8bf lo, v8bf hi) {
  return __builtin_shufflevector(lo, hi, 0,1,2,3,4,5,6,7,8,9,10,11,12,13,14,15);
}
// A-matrix 16x32 bf16 per-lane pattern: contiguous halves [2*klo..2*klo+15]
__device__ __forceinline__ v16bf ldA(const __bf16* row, int klo) {
  v8bf lo = *(const v8bf*)(row + 2*klo);
  v8bf hi = *(const v8bf*)(row + 2*klo + 8);
  return cat16(lo, hi);
}
// B-matrix: 16 contiguous halves (lane group selects K 0-15 vs 16-31)
__device__ __forceinline__ v16bf ldB16(const __bf16* p) {
  v8bf lo = *(const v8bf*)(p);
  v8bf hi = *(const v8bf*)(p + 8);
  return cat16(lo, hi);
}
__device__ __forceinline__ v16bf ldB16u(const __bf16* p) {  // possibly 2B-aligned
  v8bf_u lo = *(const v8bf_u*)(p);
  v8bf_u hi = *(const v8bf_u*)(p + 8);
  return cat16((v8bf)lo, (v8bf)hi);
}
__device__ __forceinline__ v8f wmma_bf16(v16bf a, v16bf b, v8f c) {
  return __builtin_amdgcn_wmma_f32_16x16x32_bf16(false, a, false, b, (short)0, c, false, false);
}

// ---- kernel 1: pack weights to bf16, fold softmax scale into Wq ----
__global__ void k_prep_w(const float* __restrict__ wq, const float* __restrict__ wk,
                         const float* __restrict__ wv, const float* __restrict__ wp,
                         __bf16* __restrict__ Wqkv, __bf16* __restrict__ Wpb) {
  int i = blockIdx.x*blockDim.x + threadIdx.x;
  const float scale = 0.17677669529663687f;   // HD^-0.5
  if (i < 3*CC*CC) {
    int r = i / CC;
    float v;
    if (r < CC)        v = wq[i] * scale;
    else if (r < 2*CC) v = wk[i - CC*CC];
    else               v = wv[i - 2*CC*CC];
    Wqkv[i] = (__bf16)v;
  } else {
    int j = i - 3*CC*CC;
    if (j < CC*CC) Wpb[j] = (__bf16)wp[j];
  }
}

// ---- kernel 2: LDS-tiled transpose: x (B,C,H,W) f32 -> token-major (tok,C) bf16 ----
// 32x32 tiles: coalesced 128B reads of x rows, coalesced 64B bf16 writes of tokens.
__global__ void __launch_bounds__(256)
k_x_tok(const float* __restrict__ x, __bf16* __restrict__ Xbf) {
  __shared__ __bf16 tile[32][33];             // +1 pad: no bank conflicts
  int bid = blockIdx.x;                       // BB * 8 * 288 blocks
  int b      = bid / (8*288);
  int rem    = bid % (8*288);
  int hwtile = rem >> 3;                      // 288 tiles of 32 along HW
  int ctile  = rem & 7;                       // 8 tiles of 32 along C
  int c0  = ctile*32;
  int hw0 = hwtile*32;
  int t = threadIdx.x;
  int cl  = t >> 5;                           // 0..7
  int wl  = t & 31;                           // 0..31
  #pragma unroll
  for (int r = 0; r < 4; ++r) {
    int c = cl + r*8;
    tile[c][wl] = (__bf16)x[((long)(b*CC + c0 + c))*HWSZ + hw0 + wl];
  }
  __syncthreads();
  #pragma unroll
  for (int r = 0; r < 4; ++r) {
    int hw = cl + r*8;
    Xbf[((long)(b*HWSZ + hw0 + hw))*CC + c0 + wl] = tile[wl][hw];
  }
}

// ---- kernel 3: fused QKV GEMM via WMMA, 4M x 2N register blocking ----
__global__ void __launch_bounds__(128)
k_qkv(const __bf16* __restrict__ Wqkv, const __bf16* __restrict__ Xbf,
      __bf16* __restrict__ Qb, __bf16* __restrict__ Kb, __bf16* __restrict__ Vt) {
  int wave = blockIdx.x*(blockDim.x >> 5) + (threadIdx.x >> 5);
  int lane = threadIdx.x & 31;
  int mb = wave % 12;                 // 48 M tiles / 4
  int nb = wave / 12;                 // 4608 N tiles / 2 = 2304
  int ln = lane & 15, hf = lane >> 4;
  int klo = hf ? 8 : 0, klo2 = hf ? 16 : 0;
  int mt0 = mb*4;
  long t0 = (long)nb*32;
  const __bf16* a0 = Wqkv + (long)((mt0+0)*16 + ln)*CC;
  const __bf16* a1 = Wqkv + (long)((mt0+1)*16 + ln)*CC;
  const __bf16* a2 = Wqkv + (long)((mt0+2)*16 + ln)*CC;
  const __bf16* a3 = Wqkv + (long)((mt0+3)*16 + ln)*CC;
  const __bf16* br0 = Xbf + (t0 + ln)*CC;
  const __bf16* br1 = br0 + 16*CC;
  v8f acc[4][2] = {};
  #pragma unroll
  for (int ks = 0; ks < CC; ks += 32) {
    v16bf b0 = ldB16(br0 + ks + klo2);
    v16bf b1 = ldB16(br1 + ks + klo2);
    v16bf a;
    a = ldA(a0 + ks, klo); acc[0][0] = wmma_bf16(a, b0, acc[0][0]); acc[0][1] = wmma_bf16(a, b1, acc[0][1]);
    a = ldA(a1 + ks, klo); acc[1][0] = wmma_bf16(a, b0, acc[1][0]); acc[1][1] = wmma_bf16(a, b1, acc[1][1]);
    a = ldA(a2 + ks, klo); acc[2][0] = wmma_bf16(a, b0, acc[2][0]); acc[2][1] = wmma_bf16(a, b1, acc[2][1]);
    a = ldA(a3 + ks, klo); acc[3][0] = wmma_bf16(a, b0, acc[3][0]); acc[3][1] = wmma_bf16(a, b1, acc[3][1]);
  }
  #pragma unroll
  for (int j = 0; j < 2; ++j) {
    long t = t0 + j*16 + ln;
    int b_ = (int)(t / HWSZ), hw = (int)(t % HWSZ);
    #pragma unroll
    for (int i = 0; i < 4; ++i) {
      #pragma unroll
      for (int r = 0; r < 8; ++r) {
        int Mg   = (mt0+i)*16 + r + hf*8;
        int type = Mg >> 8;             // 0=Q 1=K 2=V (uniform per i)
        int ch   = Mg & 255;
        int head = ch >> 5, d = ch & 31;
        __bf16 val = (__bf16)acc[i][j][r];
        long bh = (long)(b_*NHEADS + head);
        if (type == 0)      Qb[(bh*HWSZ + hw)*HD + d] = val;
        else if (type == 1) Kb[(bh*HWSZ + hw)*HD + d] = val;
        else                Vt[(bh*HD + d)*HWSZ + hw] = val;   // V transposed
      }
    }
  }
}

// ---- kernel 4: fused neighborhood attention (QK^T, softmax+bias, AV) ----
__global__ void __launch_bounds__(64)
k_attn(const __bf16* __restrict__ Qb, const __bf16* __restrict__ Kb,
       const __bf16* __restrict__ Vt, const float* __restrict__ rpb,
       __bf16* __restrict__ AO) {
  __shared__ float  Ssm[2][KWIN][16][32];
  __shared__ __bf16 Psm[2][KWIN][16][32];
  int wv   = threadIdx.x >> 5;
  int lane = threadIdx.x & 31;
  int wave = blockIdx.x*2 + wv;
  int wt   = wave % 6;                      // 96/16 column tiles
  int hrow = (wave/6) % HH;
  int head = (wave/(6*HH)) % NHEADS;
  int b    = wave / (6*HH*NHEADS);
  int w0   = wt*16;
  int ln = lane & 15, hf = lane >> 4;
  int klo = hf ? 8 : 0, klo2 = hf ? 16 : 0;
  int sh = min(max(hrow-3, 0), SCLIP);
  int c0 = min(max(w0-3, 0), SCLIP);        // left edge of 32-col key span
  long bh = (long)(b*NHEADS + head);

  // Q A-tile (16 queries x 32 dims), reused across all 7 window rows
  const __bf16* Qbase = Qb + (bh*HWSZ + (long)hrow*WWD + w0)*HD;
  v16bf qa = ldA(Qbase + ln*HD, klo);

  float  (*S)[16][32] = Ssm[wv];
  __bf16 (*P)[16][32] = Psm[wv];

  // S[ki][q][col] = q . k  over a 32-column span (2 WMMAs per window row)
  for (int ki = 0; ki < KWIN; ++ki) {
    const __bf16* Krow = Kb + (bh*HWSZ + (long)(sh+ki)*WWD)*HD;
    v16bf b1 = ldB16(Krow + (long)(c0 + ln)*HD + klo2);
    v16bf b2 = ldB16(Krow + (long)(c0 + 16 + ln)*HD + klo2);   // padded over-read OK
    v8f z = {};
    v8f s1 = wmma_bf16(qa, b1, z);
    v8f s2 = wmma_bf16(qa, b2, z);
    #pragma unroll
    for (int r = 0; r < 8; ++r) {
      S[ki][r + hf*8][ln]      = s1[r];
      S[ki][r + hf*8][ln + 16] = s2[r];
    }
  }
  // zero probability matrix (unused span columns must be exactly 0)
  {
    unsigned int* pz = (unsigned int*)&Psm[wv][0][0][0];
    for (int i = lane; i < (KWIN*16*32)/2; i += 32) pz[i] = 0u;
  }
  // per-query softmax over the 49 selected entries, + relative position bias
  if (lane < 16) {
    int q  = lane;
    int wq = w0 + q;
    int sw = min(max(wq-3, 0), SCLIP);
    const float* bias = rpb + ((long)head*13 + (sh - hrow + 6))*13 + (sw - wq + 6);
    float m = -1e30f;
    for (int ki = 0; ki < KWIN; ++ki)
      for (int kj = 0; kj < KWIN; ++kj)
        m = fmaxf(m, S[ki][q][sw - c0 + kj] + bias[ki*13 + kj]);
    float sum = 0.f;
    for (int ki = 0; ki < KWIN; ++ki)
      for (int kj = 0; kj < KWIN; ++kj) {
        float e = __expf(S[ki][q][sw - c0 + kj] + bias[ki*13 + kj] - m);
        S[ki][q][sw - c0 + kj] = e;
        sum += e;
      }
    float inv = 1.f / sum;
    for (int ki = 0; ki < KWIN; ++ki)
      for (int kj = 0; kj < KWIN; ++kj)
        P[ki][q][sw - c0 + kj] = (__bf16)(S[ki][q][sw - c0 + kj] * inv);
  }
  // AV: out(16q x 32d) = sum_ki P_ki(16x32cols) x V_ki(32cols x 32d)
  v8f co0 = {}, co1 = {};
  for (int ki = 0; ki < KWIN; ++ki) {
    v16bf pa = ldA(&P[ki][ln][0], klo);
    const __bf16* v0 = Vt + (bh*HD + ln     )*HWSZ + (long)(sh+ki)*WWD + c0 + klo2;
    const __bf16* v1 = Vt + (bh*HD + 16 + ln)*HWSZ + (long)(sh+ki)*WWD + c0 + klo2;
    co0 = wmma_bf16(pa, ldB16u(v0), co0);
    co1 = wmma_bf16(pa, ldB16u(v1), co1);
  }
  #pragma unroll
  for (int r = 0; r < 8; ++r) {
    int q = r + hf*8;
    long tok = (long)b*HWSZ + (long)hrow*WWD + w0 + q;
    AO[tok*CC + head*HD + ln]      = (__bf16)co0[r];
    AO[tok*CC + head*HD + 16 + ln] = (__bf16)co1[r];
  }
}

// ---- kernel 5: output projection GEMM + bias -> fp32 (B,C,H,W), 4M x 2N blocked ----
__global__ void __launch_bounds__(128)
k_proj(const __bf16* __restrict__ Wpb, const __bf16* __restrict__ AO,
       const float* __restrict__ bproj, float* __restrict__ out) {
  int wave = blockIdx.x*(blockDim.x >> 5) + (threadIdx.x >> 5);
  int lane = threadIdx.x & 31;
  int mb = wave & 3;                  // 16 M tiles / 4
  int nb = wave >> 2;                 // 2304 N tile pairs
  int ln = lane & 15, hf = lane >> 4;
  int klo = hf ? 8 : 0, klo2 = hf ? 16 : 0;
  int mt0 = mb*4;
  long t0 = (long)nb*32;
  const __bf16* a0 = Wpb + (long)((mt0+0)*16 + ln)*CC;
  const __bf16* a1 = Wpb + (long)((mt0+1)*16 + ln)*CC;
  const __bf16* a2 = Wpb + (long)((mt0+2)*16 + ln)*CC;
  const __bf16* a3 = Wpb + (long)((mt0+3)*16 + ln)*CC;
  const __bf16* br0 = AO + (t0 + ln)*CC;
  const __bf16* br1 = br0 + 16*CC;
  v8f acc[4][2] = {};
  #pragma unroll
  for (int ks = 0; ks < CC; ks += 32) {
    v16bf b0 = ldB16(br0 + ks + klo2);
    v16bf b1 = ldB16(br1 + ks + klo2);
    v16bf a;
    a = ldA(a0 + ks, klo); acc[0][0] = wmma_bf16(a, b0, acc[0][0]); acc[0][1] = wmma_bf16(a, b1, acc[0][1]);
    a = ldA(a1 + ks, klo); acc[1][0] = wmma_bf16(a, b0, acc[1][0]); acc[1][1] = wmma_bf16(a, b1, acc[1][1]);
    a = ldA(a2 + ks, klo); acc[2][0] = wmma_bf16(a, b0, acc[2][0]); acc[2][1] = wmma_bf16(a, b1, acc[2][1]);
    a = ldA(a3 + ks, klo); acc[3][0] = wmma_bf16(a, b0, acc[3][0]); acc[3][1] = wmma_bf16(a, b1, acc[3][1]);
  }
  #pragma unroll
  for (int j = 0; j < 2; ++j) {
    long t = t0 + j*16 + ln;
    int b_ = (int)(t / HWSZ), hw = (int)(t % HWSZ);
    #pragma unroll
    for (int i = 0; i < 4; ++i) {
      #pragma unroll
      for (int r = 0; r < 8; ++r) {
        int o = (mt0+i)*16 + r + hf*8;
        out[((long)(b_*CC + o))*HWSZ + hw] = acc[i][j][r] + bproj[o];
      }
    }
  }
}

extern "C" void kernel_launch(void* const* d_in, const int* in_sizes, int n_in,
                              void* d_out, int out_size, void* d_ws, size_t ws_size,
                              hipStream_t stream) {
  const float* x   = (const float*)d_in[0];
  const float* wq  = (const float*)d_in[1];
  const float* wk  = (const float*)d_in[2];
  const float* wv  = (const float*)d_in[3];
  const float* wp  = (const float*)d_in[4];
  const float* bp  = (const float*)d_in[5];
  const float* rpb = (const float*)d_in[6];
  (void)in_sizes; (void)n_in; (void)out_size; (void)ws_size;

  char* ws = (char*)d_ws;
  size_t off = 0;
  auto alloc = [&](size_t bytes) { size_t o = off; off += (bytes + 255) & ~(size_t)255; return o; };
  __bf16* Wqkv = (__bf16*)(ws + alloc((size_t)3*CC*CC*2));
  __bf16* Wpb  = (__bf16*)(ws + alloc((size_t)CC*CC*2));
  __bf16* Xbf  = (__bf16*)(ws + alloc((size_t)NTOK*CC*2));
  __bf16* Qb   = (__bf16*)(ws + alloc((size_t)NTOK*CC*2 + 8192));  // +pad for span over-read
  __bf16* Kb   = (__bf16*)(ws + alloc((size_t)NTOK*CC*2 + 8192));
  __bf16* Vt   = (__bf16*)(ws + alloc((size_t)NTOK*CC*2 + 8192));
  __bf16* AO   = Xbf;                 // alias: Xbf dead after k_qkv (stream-ordered)
  float*  out  = (float*)d_out;

  k_prep_w<<<(4*CC*CC + 255)/256, 256, 0, stream>>>(wq, wk, wv, wp, Wqkv, Wpb);
  k_x_tok <<<BB*8*288,            256, 0, stream>>>(x, Xbf);
  k_qkv   <<<(12*2304)/4,         128, 0, stream>>>(Wqkv, Xbf, Qb, Kb, Vt);
  k_attn  <<<(BB*NHEADS*HH*6)/2,   64, 0, stream>>>(Qb, Kb, Vt, rpb, AO);
  k_proj  <<<(4*2304)/4,          128, 0, stream>>>(Wpb, AO, bp, out);
}
